// FlashAttention_33457795236112
// MI455X (gfx1250) — matrix operand order repaired and checked by reading, not compiled, not run on hardware
//
#include <hip/hip_runtime.h>

// ---------------------------------------------------------------------------
// MHA block for MI455X (gfx1250): v_wmma_f32_16x16x32_f16 everywhere,
// async global->LDS staging (ASYNCcnt) for flash-attention K/V tiles,
// register-double-buffered (software pipelined) GEMM inner loop.
// ---------------------------------------------------------------------------

typedef __attribute__((ext_vector_type(16))) _Float16 v16h;
typedef __attribute__((ext_vector_type(8)))  _Float16 v8h;
typedef __attribute__((ext_vector_type(8)))  float    v8f;

#define EMBED   1024
#define HEADS   16
#define HDIM    64
#define SEQ     2048
#define BATCH   4
#define ROWS    (BATCH * SEQ)          // 8192
#define ATT_SCALE 0.125f               // 64^-0.5
#define FA_WAVES 8                     // waves per flash block
#define CHUNK   32                     // keys per chunk

__device__ __forceinline__ v8f wmma_f16(v16h a, v16h b, v8f c) {
  return __builtin_amdgcn_wmma_f32_16x16x32_f16(
      false, a, false, b, (short)0, c, false, false);
}

// CDNA5 async copy: global -> LDS, 16B per enabled lane, tracked by ASYNCcnt.
// LDS byte address = low 32 bits of the generic pointer (ISA 10.2 aperture).
__device__ __forceinline__ void async_b128(void* dst_lds, const void* src) {
  const unsigned l = (unsigned)(uintptr_t)dst_lds;
  const unsigned long long g = (unsigned long long)(uintptr_t)src;
  asm volatile("global_load_async_to_lds_b128 %0, %1, off"
               :: "v"(l), "v"(g) : "memory");
}
__device__ __forceinline__ void wait_async0() {
  asm volatile("s_wait_asynccnt 0" ::: "memory");
}

// A-fragment (16x32 f16) from row-major [M][ld]:
// lane m = lane&15; g = lane>>4; elems 0..15 -> K=16g+0..15
__device__ __forceinline__ v16h load_a_rm(const _Float16* base, int ld, int m0,
                                          int k0, int lane) {
  const int m = lane & 15, g = lane >> 4;
  const _Float16* p = base + (size_t)(m0 + m) * ld + k0 + 16 * g;
  union { v16h v; v8h h[2]; } u;
  u.h[0] = *(const v8h*)(p);
  u.h[1] = *(const v8h*)(p + 8);
  return u.v;
}

// A-fragment from a 16x32 f16 LDS tile (row stride 32 halves)
__device__ __forceinline__ v16h load_a_lds(const _Float16* p0, int lane) {
  const int m = lane & 15, g = lane >> 4;
  const _Float16* p = p0 + m * 32 + 16 * g;
  union { v16h v; v8h h[2]; } u;
  u.h[0] = *(const v8h*)(p);
  u.h[1] = *(const v8h*)(p + 8);
  return u.v;
}

// B-fragment (32x16 f16), B[k][n] = src[(n0+n)*ld + k0 + k]:
// lane col n = lane&15; K elem j = 16*(lane>>4)+j -> one contiguous 32B load.
__device__ __forceinline__ v16h load_b_tr(const _Float16* base, int ld, int n0,
                                          int k0, int lane) {
  const int n = lane & 15, g = lane >> 4;
  return *(const v16h*)(base + (size_t)(n0 + n) * ld + k0 + 16 * g);
}

// ---------------------------------------------------------------------------
__global__ void k_cvt(const float* __restrict__ in, _Float16* __restrict__ out,
                      int n) {
  const int i4 = (blockIdx.x * blockDim.x + threadIdx.x) * 4;
  if (i4 + 3 < n) {
    float4 v = *(const float4*)(in + i4);
    out[i4 + 0] = (_Float16)v.x;
    out[i4 + 1] = (_Float16)v.y;
    out[i4 + 2] = (_Float16)v.z;
    out[i4 + 3] = (_Float16)v.w;
  }
}

__global__ void k_transpose_cvt(const float* __restrict__ W,
                                _Float16* __restrict__ Wt) {
  const int n = blockIdx.x * 16 + threadIdx.x;
  const int k = blockIdx.y * 16 + threadIdx.y;
  Wt[(size_t)n * EMBED + k] = (_Float16)W[(size_t)k * EMBED + n];
}

// ---------------------------------------------------------------------------
// GEMM: Y[8192][1024] = A @ W + b (W transposed as Wt). Wave computes a
// 16x64 strip; A-fragment reused across 4 B-fragments; fragments for step
// k+1 are prefetched into registers before the 4 WMMAs of step k.
// mode 0: f16 out [B,H,T,D] (Q,K) | mode 1: f16 out [B,H,D,T] (V) |
// mode 2: f32 out [B,T,C] (final projection)
__global__ void k_gemm(const _Float16* __restrict__ A,
                       const _Float16* __restrict__ Wt,
                       const float* __restrict__ bias,
                       _Float16* __restrict__ out16,
                       float* __restrict__ out32, int mode) {
  const int lane = threadIdx.x & 31;
  const int wave = threadIdx.x >> 5;                 // 4 waves / block
  const int m0 = blockIdx.x * 16;
  const int n0 = blockIdx.y * 256 + wave * 64;

  v8f acc[4] = {{}, {}, {}, {}};
  v16h a  = load_a_rm(A, EMBED, m0, 0, lane);
  v16h b0 = load_b_tr(Wt, EMBED, n0 +  0, 0, lane);
  v16h b1 = load_b_tr(Wt, EMBED, n0 + 16, 0, lane);
  v16h b2 = load_b_tr(Wt, EMBED, n0 + 32, 0, lane);
  v16h b3 = load_b_tr(Wt, EMBED, n0 + 48, 0, lane);

#pragma unroll 2
  for (int k0 = 32; k0 < EMBED; k0 += 32) {
    // prefetch step k0 while computing on step k0-32
    const v16h an  = load_a_rm(A, EMBED, m0, k0, lane);
    const v16h bn0 = load_b_tr(Wt, EMBED, n0 +  0, k0, lane);
    const v16h bn1 = load_b_tr(Wt, EMBED, n0 + 16, k0, lane);
    const v16h bn2 = load_b_tr(Wt, EMBED, n0 + 32, k0, lane);
    const v16h bn3 = load_b_tr(Wt, EMBED, n0 + 48, k0, lane);
    acc[0] = wmma_f16(a, b0, acc[0]);
    acc[1] = wmma_f16(a, b1, acc[1]);
    acc[2] = wmma_f16(a, b2, acc[2]);
    acc[3] = wmma_f16(a, b3, acc[3]);
    a = an; b0 = bn0; b1 = bn1; b2 = bn2; b3 = bn3;
  }
  acc[0] = wmma_f16(a, b0, acc[0]);
  acc[1] = wmma_f16(a, b1, acc[1]);
  acc[2] = wmma_f16(a, b2, acc[2]);
  acc[3] = wmma_f16(a, b3, acc[3]);

  const int g = lane >> 4, nn = lane & 15;
#pragma unroll
  for (int j = 0; j < 4; ++j) {
    const int col = n0 + 16 * j + nn;
    const float bv = bias[col];
#pragma unroll
    for (int r = 0; r < 8; ++r) {
      const int row = m0 + r + 8 * g;
      const float val = acc[j][r] + bv;
      if (mode == 2) {
        out32[(size_t)row * EMBED + col] = val;
      } else {
        const int bb = row >> 11, t = row & (SEQ - 1);
        const int h = col >> 6, d = col & 63;
        const size_t hb = (size_t)(bb * HEADS + h) * SEQ * HDIM;
        const size_t idx = (mode == 0) ? hb + (size_t)t * HDIM + d
                                       : hb + (size_t)d * SEQ + t;
        out16[idx] = (_Float16)val;
      }
    }
  }
}

// ---------------------------------------------------------------------------
// Flash attention, block = 8 waves = 128 queries sharing one (b,h).
// K/V chunks double-buffered in LDS via GLOBAL_LOAD_ASYNC_TO_LDS_B128:
// next chunk's copies are issued right after the barrier, overlapping the
// current chunk's WMMAs; s_wait_asynccnt 0 at loop top drains them.
__global__ void k_flash(const _Float16* __restrict__ Q,   // [BH][T][D]
                        const _Float16* __restrict__ K,   // [BH][T][D]
                        const _Float16* __restrict__ Vt,  // [BH][D][T]
                        _Float16* __restrict__ O) {       // [B][T][C] f16
  __shared__ __align__(32) _Float16 kt[2][CHUNK * HDIM];     // 2 x 4KB
  __shared__ __align__(32) _Float16 vtl[2][HDIM * CHUNK];    // 2 x 4KB
  __shared__ __align__(32) _Float16 pt[FA_WAVES][16 * 32];   // 8KB
  const int tid = threadIdx.x;
  const int lane = tid & 31;
  const int wave = tid >> 5;
  const int bh = blockIdx.y;
  const int q0 = blockIdx.x * (FA_WAVES * 16) + wave * 16;
  const size_t hb = (size_t)bh * SEQ * HDIM;
  const _Float16* Qh = Q + hb;
  const _Float16* Kh = K + hb;
  const _Float16* Vh = Vt + hb;

  const v16h qa0 = load_a_rm(Qh, HDIM, q0, 0, lane);
  const v16h qa1 = load_a_rm(Qh, HDIM, q0, 32, lane);

  v8f o0 = {}, o1 = {}, o2 = {}, o3 = {};
  float m_i[8], l_i[8];
#pragma unroll
  for (int r = 0; r < 8; ++r) { m_i[r] = -1e30f; l_i[r] = 0.0f; }

  const int vrow = tid >> 2, vq = tid & 3;  // V copy: 64 rows x 4 x 16B
  auto issue_chunk = [&](int buf, int t0) {
    // K chunk: 32 rows x 64 halves = 4KB contiguous; 256 threads x 16B
    async_b128(&kt[buf][tid * 8], Kh + (size_t)t0 * HDIM + tid * 8);
    // V chunk: 64 d-rows x 32 keys, row stride SEQ in global
    async_b128(&vtl[buf][vrow * CHUNK + vq * 8],
               Vh + (size_t)vrow * SEQ + t0 + vq * 8);
  };
  issue_chunk(0, 0);

  _Float16* pl = pt[wave];
  const int g = lane >> 4, nn = lane & 15;

  for (int t0 = 0; t0 < SEQ; t0 += CHUNK) {
    const int cur = (t0 >> 5) & 1;
    wait_async0();        // this wave's copies for `cur` have landed
    __syncthreads();      // all waves' copies have landed
    if (t0 + CHUNK < SEQ) issue_chunk(cur ^ 1, t0 + CHUNK);  // overlap next

    const _Float16* kl = kt[cur];
    v8f s0 = {}, s1 = {};
    s0 = wmma_f16(qa0, load_b_tr(kl, HDIM, 0, 0, lane), s0);
    s0 = wmma_f16(qa1, load_b_tr(kl, HDIM, 0, 32, lane), s0);
    s1 = wmma_f16(qa0, load_b_tr(kl, HDIM, 16, 0, lane), s1);
    s1 = wmma_f16(qa1, load_b_tr(kl, HDIM, 16, 32, lane), s1);

#pragma unroll
    for (int r = 0; r < 8; ++r) {
      const float a = s0[r] * ATT_SCALE;
      const float b = s1[r] * ATT_SCALE;
      float mx = fmaxf(a, b);                 // row r+8g across 16 lanes
      mx = fmaxf(mx, __shfl_xor(mx, 1));
      mx = fmaxf(mx, __shfl_xor(mx, 2));
      mx = fmaxf(mx, __shfl_xor(mx, 4));
      mx = fmaxf(mx, __shfl_xor(mx, 8));
      const float mnew = fmaxf(m_i[r], mx);
      const float corr = __expf(m_i[r] - mnew);
      const float p0 = __expf(a - mnew);
      const float p1 = __expf(b - mnew);
      float rs = p0 + p1;
      rs += __shfl_xor(rs, 1);
      rs += __shfl_xor(rs, 2);
      rs += __shfl_xor(rs, 4);
      rs += __shfl_xor(rs, 8);
      l_i[r] = l_i[r] * corr + rs;
      m_i[r] = mnew;
      o0[r] *= corr; o1[r] *= corr; o2[r] *= corr; o3[r] *= corr;
      const int ro = (r + 8 * g) * 32 + nn;
      pl[ro]      = (_Float16)p0;
      pl[ro + 16] = (_Float16)p1;
    }
    __syncthreads();

    const v16h pa = load_a_lds(pl, lane);
    const _Float16* vl = vtl[cur];
    o0 = wmma_f16(pa, load_b_tr(vl, CHUNK, 0,  0, lane), o0);
    o1 = wmma_f16(pa, load_b_tr(vl, CHUNK, 16, 0, lane), o1);
    o2 = wmma_f16(pa, load_b_tr(vl, CHUNK, 32, 0, lane), o2);
    o3 = wmma_f16(pa, load_b_tr(vl, CHUNK, 48, 0, lane), o3);
    __syncthreads();      // everyone done with `cur` before it is refilled
  }

  const int bb = bh >> 4, h = bh & 15;
#pragma unroll
  for (int r = 0; r < 8; ++r) {
    const float inv = 1.0f / l_i[r];
    const int t = q0 + r + 8 * g;
    const size_t base = ((size_t)bb * SEQ + t) * EMBED + h * HDIM + nn;
    O[base]      = (_Float16)(o0[r] * inv);
    O[base + 16] = (_Float16)(o1[r] * inv);
    O[base + 32] = (_Float16)(o2[r] * inv);
    O[base + 48] = (_Float16)(o3[r] * inv);
  }
}

// ---------------------------------------------------------------------------
extern "C" void kernel_launch(void* const* d_in, const int* in_sizes, int n_in,
                              void* d_out, int out_size, void* d_ws,
                              size_t ws_size, hipStream_t stream) {
  (void)in_sizes; (void)n_in; (void)out_size; (void)ws_size;
  const float* x  = (const float*)d_in[0];
  const float* Wq = (const float*)d_in[1];
  const float* bq = (const float*)d_in[2];
  const float* Wk = (const float*)d_in[3];
  const float* bk = (const float*)d_in[4];
  const float* Wv = (const float*)d_in[5];
  const float* bv = (const float*)d_in[6];
  const float* Wo = (const float*)d_in[7];
  const float* bo = (const float*)d_in[8];

  char* ws = (char*)d_ws;
  const size_t MB = 1u << 20;
  _Float16* xh  = (_Float16*)(ws + 0 * MB);
  _Float16* Wqt = (_Float16*)(ws + 16 * MB);
  _Float16* Wkt = (_Float16*)(ws + 18 * MB);
  _Float16* Wvt = (_Float16*)(ws + 20 * MB);
  _Float16* Wot = (_Float16*)(ws + 22 * MB);
  _Float16* Qh  = (_Float16*)(ws + 24 * MB);
  _Float16* Kh  = (_Float16*)(ws + 40 * MB);
  _Float16* Vh  = (_Float16*)(ws + 56 * MB);
  _Float16* Oh  = (_Float16*)(ws + 72 * MB);

  const int NTOT = ROWS * EMBED;
  k_cvt<<<NTOT / (256 * 4), 256, 0, stream>>>(x, xh, NTOT);

  dim3 tb(16, 16), tg(EMBED / 16, EMBED / 16);
  k_transpose_cvt<<<tg, tb, 0, stream>>>(Wq, Wqt);
  k_transpose_cvt<<<tg, tb, 0, stream>>>(Wk, Wkt);
  k_transpose_cvt<<<tg, tb, 0, stream>>>(Wv, Wvt);
  k_transpose_cvt<<<tg, tb, 0, stream>>>(Wo, Wot);

  dim3 gg(ROWS / 16, EMBED / 256);  // (512, 4), 4 waves/block, 16x64 strips
  k_gemm<<<gg, 128, 0, stream>>>(xh, Wqt, bq, Qh, nullptr, 0);
  k_gemm<<<gg, 128, 0, stream>>>(xh, Wkt, bk, Kh, nullptr, 0);
  k_gemm<<<gg, 128, 0, stream>>>(xh, Wvt, bv, Vh, nullptr, 1);

  dim3 fg(SEQ / (FA_WAVES * 16), BATCH * HEADS);  // (16, 64), 256 thr/block
  k_flash<<<fg, 256, 0, stream>>>(Qh, Kh, Vh, Oh);

  k_gemm<<<gg, 128, 0, stream>>>(Oh, Wot, bo, nullptr, (float*)d_out, 2);
}
